// JointNet_66511863545982
// MI455X (gfx1250) — matrix operand order repaired and checked by reading, not compiled, not run on hardware
//
#include <hip/hip_runtime.h>
#include <hip/hip_bf16.h>
#include <math.h>

typedef __attribute__((ext_vector_type(16))) __bf16 v16bf;
typedef __attribute__((ext_vector_type(8)))  float  v8f;

#define B_  8
#define T_  200
#define U_  50
#define DE  512
#define DJ  1024
#define V_  1024

#define WMMA_BF16(a, b, c) \
  __builtin_amdgcn_wmma_f32_16x16x32_bf16(false, (a), false, (b), (short)0, (c), false, false)

static __device__ __forceinline__ unsigned short f2bf(float f) {
  // round-to-nearest-even float32 -> bf16 (inputs are well-behaved: tanh/uniform)
  unsigned u = __builtin_bit_cast(unsigned, f);
  u += 0x7FFFu + ((u >> 16) & 1u);
  return (unsigned short)(u >> 16);
}

static __device__ __forceinline__ float fast_tanh(float x) {
  // CDNA5 transcendental unit tanh (TRANS32 op per ISA §8.4 list)
  float r;
  asm volatile("v_tanh_f32 %0, %1" : "=v"(r) : "v"(x));
  return r;
}

// ---------------------------------------------------------------------------
// Kernel 1: enc = encoder_output @ W_enc^T + b_enc ; dec = decoder @ W_dec^T + b_dec
// One block per output row (1600 enc rows + 400 dec rows), fp32.
// ---------------------------------------------------------------------------
__global__ __launch_bounds__(256) void proj_kernel(
    const float* __restrict__ encIn, const float* __restrict__ decIn,
    const float* __restrict__ W_enc, const float* __restrict__ b_enc,
    const float* __restrict__ W_dec, const float* __restrict__ b_dec,
    float* __restrict__ encOut, float* __restrict__ decOut) {
  __shared__ float xs[DE];
  const int row = blockIdx.x;
  const float* x; const float* W; const float* bias; float* outp;
  if (row < B_ * T_) {
    x = encIn + (size_t)row * DE; W = W_enc; bias = b_enc;
    outp = encOut + (size_t)row * DJ;
  } else {
    const int rr = row - B_ * T_;
    x = decIn + (size_t)rr * DE; W = W_dec; bias = b_dec;
    outp = decOut + (size_t)rr * DJ;
  }
  for (int k = threadIdx.x; k < DE; k += 256) xs[k] = x[k];
  __syncthreads();
  for (int j = threadIdx.x; j < DJ; j += 256) {
    const float4* wr = (const float4*)(W + (size_t)j * DE);
    float acc = 0.f;
#pragma unroll 4
    for (int k4 = 0; k4 < DE / 4; ++k4) {
      float4 w = wr[k4];
      acc += w.x * xs[4 * k4 + 0] + w.y * xs[4 * k4 + 1] +
             w.z * xs[4 * k4 + 2] + w.w * xs[4 * k4 + 3];
    }
    outp[j] = acc + bias[j];
  }
}

// ---------------------------------------------------------------------------
// Kernel 2: pack W_fc (V x DJ, row-major fp32) into bf16 WMMA B-operand tiles.
// Packed index p = ((nt*32 + ks)*32 + lane)*16 + h
//   column n = nt*16 + (lane&15), k = ks*32 + ((lane>>4)<<4) + h
// (B 32x16 bf16: lanes 0-15 hold K=0..15 of col n, lanes 16-31 hold K=16..31)
// ---------------------------------------------------------------------------
__global__ __launch_bounds__(256) void pack_wfc_kernel(
    const float* __restrict__ W_fc, unsigned short* __restrict__ Bpk) {
  const int p    = blockIdx.x * 256 + threadIdx.x;  // 0 .. 1M-1
  const int h    = p & 15;
  const int lane = (p >> 4) & 31;
  const int ks   = (p >> 9) & 31;
  const int nt   = p >> 14;
  const int n = nt * 16 + (lane & 15);
  const int k = ks * 32 + ((lane >> 4) << 4) + h;
  Bpk[p] = f2bf(W_fc[(size_t)n * DJ + k]);
}

// ---------------------------------------------------------------------------
// Kernel 3: per (b,t): A = tanh(enc[b,t] + dec[b,u]) (u=0..49, padded to 64),
// packed bf16 in LDS in WMMA A layout. 8 waves; each wave owns 8 N-tiles,
// processed 2 at a time so 4 LDS A-fragments feed 8 WMMAs per K-step.
// ---------------------------------------------------------------------------
__global__ __launch_bounds__(256) void joint_gemm_kernel(
    const float* __restrict__ encP, const float* __restrict__ decP,
    const unsigned short* __restrict__ Bpk, const float* __restrict__ b_fc,
    float* __restrict__ out) {
  const int bt = blockIdx.x;            // 0..1599  (b*200 + t)
  const int b  = bt / T_;
  __shared__ float encRow[DJ];
  __shared__ __align__(32) unsigned short Apk[4 * 32 * 32 * 16];  // 128 KB

  for (int k = threadIdx.x; k < DJ; k += 256)
    encRow[k] = encP[(size_t)bt * DJ + k];
  __syncthreads();

  // Phase 1: build packed A tiles (rows 50..63 are zero padding).
  for (int idx = threadIdx.x; idx < 64 * DJ; idx += 256) {
    const int m = idx >> 10;
    const int k = idx & (DJ - 1);
    float v = 0.f;
    if (m < U_) v = fast_tanh(encRow[k] + decP[((size_t)(b * U_ + m)) * DJ + k]);
    const int mt = m >> 4, mr = m & 15, ks = k >> 5, kk = k & 31;
    const int lane = mr + ((kk >> 4) << 4);
    const int h    = kk & 15;
    Apk[(((mt * 32 + ks) * 32) + lane) * 16 + h] = f2bf(v);
  }
  __syncthreads();

  // Phase 2: WMMA accumulation. 2 N-tiles x 4 M-tiles per pass.
  const int wave = threadIdx.x >> 5;
  const int lane = threadIdx.x & 31;
  const int mrow = (lane >> 4) << 3;    // 0 or 8 (C layout M offset)
  const int ncol = lane & 15;           // C layout N = lane&15
  const v16bf* ap = (const v16bf*)Apk;

#pragma unroll 1
  for (int nti = 0; nti < 4; ++nti) {
    const int nt0 = (wave << 3) + (nti << 1);
    const int nt1 = nt0 + 1;
    v8f c00 = {}, c01 = {}, c02 = {}, c03 = {};
    v8f c10 = {}, c11 = {}, c12 = {}, c13 = {};
    const v16bf* bp0 = (const v16bf*)Bpk + ((size_t)nt0 * 32 * 32 + lane);
    const v16bf* bp1 = (const v16bf*)Bpk + ((size_t)nt1 * 32 * 32 + lane);
    // unroll 1: keep loads adjacent to WMMAs, avoid hoist-then-spill.
#pragma unroll 1
    for (int ks = 0; ks < 32; ++ks) {
      v16bf b0 = bp0[ks * 32];
      v16bf b1 = bp1[ks * 32];
      v16bf a0 = ap[(0 * 32 + ks) * 32 + lane];
      v16bf a1 = ap[(1 * 32 + ks) * 32 + lane];
      v16bf a2 = ap[(2 * 32 + ks) * 32 + lane];
      v16bf a3 = ap[(3 * 32 + ks) * 32 + lane];
      c00 = WMMA_BF16(a0, b0, c00);
      c10 = WMMA_BF16(a0, b1, c10);
      c01 = WMMA_BF16(a1, b0, c01);
      c11 = WMMA_BF16(a1, b1, c11);
      c02 = WMMA_BF16(a2, b0, c02);
      c12 = WMMA_BF16(a2, b1, c12);
      c03 = WMMA_BF16(a3, b0, c03);
      c13 = WMMA_BF16(a3, b1, c13);
    }
    const float bias0 = b_fc[(nt0 << 4) + ncol];
    const float bias1 = b_fc[(nt1 << 4) + ncol];
    float* o0 = out + ((size_t)bt * U_) * V_ + (nt0 << 4) + ncol;
    float* o1 = o0 + 16;
#pragma unroll
    for (int r = 0; r < 8; ++r) {
      const int u0 = 0  + mrow + r;
      const int u1 = 16 + mrow + r;
      const int u2 = 32 + mrow + r;
      const int u3 = 48 + mrow + r;
      o0[(size_t)u0 * V_] = c00[r] + bias0;
      o0[(size_t)u1 * V_] = c01[r] + bias0;
      o0[(size_t)u2 * V_] = c02[r] + bias0;
      if (u3 < U_) o0[(size_t)u3 * V_] = c03[r] + bias0;
      o1[(size_t)u0 * V_] = c10[r] + bias1;
      o1[(size_t)u1 * V_] = c11[r] + bias1;
      o1[(size_t)u2 * V_] = c12[r] + bias1;
      if (u3 < U_) o1[(size_t)u3 * V_] = c13[r] + bias1;
    }
  }
}

// ---------------------------------------------------------------------------
// Kernel 4: in-place log_softmax over last dim (1024). One block per row.
// ---------------------------------------------------------------------------
__global__ __launch_bounds__(256) void logsoftmax_kernel(float* __restrict__ out) {
  const size_t row = blockIdx.x;
  float* p = out + row * (size_t)V_;
  float4 x = ((const float4*)p)[threadIdx.x];

  __shared__ float redm[8];
  __shared__ float reds[8];
  const int wave = threadIdx.x >> 5;
  const int lane = threadIdx.x & 31;

  float m = fmaxf(fmaxf(x.x, x.y), fmaxf(x.z, x.w));
  for (int o = 16; o > 0; o >>= 1) m = fmaxf(m, __shfl_xor(m, o, 32));
  if (lane == 0) redm[wave] = m;
  __syncthreads();
  m = redm[0];
#pragma unroll
  for (int w = 1; w < 8; ++w) m = fmaxf(m, redm[w]);

  float s = __expf(x.x - m) + __expf(x.y - m) + __expf(x.z - m) + __expf(x.w - m);
  for (int o = 16; o > 0; o >>= 1) s += __shfl_xor(s, o, 32);
  if (lane == 0) reds[wave] = s;
  __syncthreads();
  s = reds[0];
#pragma unroll
  for (int w = 1; w < 8; ++w) s += reds[w];

  const float l = m + __logf(s);
  float4 y = make_float4(x.x - l, x.y - l, x.z - l, x.w - l);
  ((float4*)p)[threadIdx.x] = y;
}

// ---------------------------------------------------------------------------
extern "C" void kernel_launch(void* const* d_in, const int* in_sizes, int n_in,
                              void* d_out, int out_size, void* d_ws, size_t ws_size,
                              hipStream_t stream) {
  const float* encIn = (const float*)d_in[0];
  const float* decIn = (const float*)d_in[1];
  const float* W_enc = (const float*)d_in[2];
  const float* b_enc = (const float*)d_in[3];
  const float* W_dec = (const float*)d_in[4];
  const float* b_dec = (const float*)d_in[5];
  const float* W_fc  = (const float*)d_in[6];
  const float* b_fc  = (const float*)d_in[7];
  float* out = (float*)d_out;

  float* encP = (float*)d_ws;                                  // 1600*1024 fp32
  float* decP = encP + (size_t)B_ * T_ * DJ;                   //  400*1024 fp32
  unsigned short* Bpk = (unsigned short*)(decP + (size_t)B_ * U_ * DJ);  // 1M bf16

  proj_kernel<<<B_ * T_ + B_ * U_, 256, 0, stream>>>(
      encIn, decIn, W_enc, b_enc, W_dec, b_dec, encP, decP);
  pack_wfc_kernel<<<(V_ * DJ) / 256, 256, 0, stream>>>(W_fc, Bpk);
  joint_gemm_kernel<<<B_ * T_, 256, 0, stream>>>(encP, decP, Bpk, b_fc, out);
  logsoftmax_kernel<<<B_ * T_ * U_, 256, 0, stream>>>(out);
}
